// StyleGAN2Generator_76338748719242
// MI455X (gfx1250) — matrix operand order repaired and checked by reading, not compiled, not run on hardware
//
#include <hip/hip_runtime.h>
#include <hip/hip_bf16.h>
#include <math.h>

typedef __attribute__((ext_vector_type(16))) _Float16 v16h;
typedef __attribute__((ext_vector_type(8)))  _Float16 v8h;
typedef __attribute__((ext_vector_type(8)))  float    v8f;

// ---------------------------------------------------------------------------
// Mapping network: 8 layers of (8x512)@(512x512)^T + bias, batchnorm(B=8), lrelu
// ---------------------------------------------------------------------------
__global__ void __launch_bounds__(512)
mapping_kernel(const float* __restrict__ z, const int* __restrict__ labels,
               const float* __restrict__ emb, const float* __restrict__ W,
               const float* __restrict__ bias, const float* __restrict__ g,
               const float* __restrict__ beta, float* __restrict__ wout)
{
    __shared__ float xb[8][512];
    const int t = threadIdx.x;
    for (int b = 0; b < 8; ++b)
        xb[b][t] = z[b * 512 + t] + emb[labels[b] * 512 + t];
    __syncthreads();

    for (int layer = 0; layer < 8; ++layer) {
        float acc[8];
        const float bv = bias[layer * 512 + t];
#pragma unroll
        for (int b = 0; b < 8; ++b) acc[b] = bv;
        const float* wrow = &W[((size_t)layer * 512 + t) * 512];
        for (int k = 0; k < 512; ++k) {
            const float wv = wrow[k];
#pragma unroll
            for (int b = 0; b < 8; ++b) acc[b] += xb[b][k] * wv;
        }
        float mu = 0.f;
#pragma unroll
        for (int b = 0; b < 8; ++b) mu += acc[b];
        mu *= 0.125f;
        float var = 0.f;
#pragma unroll
        for (int b = 0; b < 8; ++b) { const float dd = acc[b] - mu; var += dd * dd; }
        var *= 0.125f;
        const float inv = rsqrtf(var + 1e-5f);
        const float gv = g[layer * 512 + t];
        const float be = beta[layer * 512 + t];
        __syncthreads();
#pragma unroll
        for (int b = 0; b < 8; ++b) {
            float v = (acc[b] - mu) * inv * gv + be;
            v = (v >= 0.f) ? v : 0.2f * v;
            xb[b][t] = v;
        }
        __syncthreads();
    }
    for (int b = 0; b < 8; ++b) wout[b * 512 + t] = xb[b][t];
}

// ---------------------------------------------------------------------------
// Style: s[b,cin] = dot(w[b,:512], m[cin,:512])
// ---------------------------------------------------------------------------
__global__ void style_kernel(const float* __restrict__ w, const float* __restrict__ m,
                             float* __restrict__ s, int Cin)
{
    const int b = blockIdx.y;
    const int cin = blockIdx.x * blockDim.x + threadIdx.x;
    if (cin >= Cin) return;
    const float* wb = &w[b * 512];
    const float* mr = &m[(size_t)cin * 512];
    float acc = 0.f;
    for (int k = 0; k < 512; ++k) acc += wb[k] * mr[k];
    s[b * Cin + cin] = acc;
}

// ---------------------------------------------------------------------------
// Demod: d[b,cout] = rsqrt(scale^2 * sum_cin s^2 * sum_kk weight^2 + 1e-8)
// ---------------------------------------------------------------------------
__global__ void __launch_bounds__(64)
demod_kernel(const float* __restrict__ s, const float* __restrict__ weight,
             float* __restrict__ d, int Cin, int Cout, float scale)
{
    const int b = blockIdx.y, cout = blockIdx.x;
    __shared__ float red[64];
    float acc = 0.f;
    for (int cin = threadIdx.x; cin < Cin; cin += 64) {
        const float* wp = &weight[((size_t)cout * Cin + cin) * 9];
        float wsq = 0.f;
#pragma unroll
        for (int t = 0; t < 9; ++t) wsq += wp[t] * wp[t];
        const float sv = s[b * Cin + cin];
        acc += sv * sv * wsq;
    }
    red[threadIdx.x] = acc;
    __syncthreads();
    for (int off = 32; off > 0; off >>= 1) {
        if (threadIdx.x < off) red[threadIdx.x] += red[threadIdx.x + off];
        __syncthreads();
    }
    if (threadIdx.x == 0)
        d[b * Cout + cout] = rsqrtf(scale * scale * red[0] + 1e-8f);
}

// ---------------------------------------------------------------------------
// CHW f32 -> HWC f16 (one-time transpose of the learned constant input)
// ---------------------------------------------------------------------------
__global__ void chw_to_hwc_f16(const float* __restrict__ in, _Float16* __restrict__ out,
                               int C, int HW)
{
    const int i = blockIdx.x * 256 + threadIdx.x;
    if (i >= C * HW) return;
    const int p = i / C;
    const int c = i - p * C;
    out[i] = (_Float16)in[(size_t)c * HW + p];
}

// ---------------------------------------------------------------------------
// Precompute style-modulated weights in WMMA A-fragment order (f16):
//   wfrag[((b*CT + ct)*NC + kc)*512 + l*16 + j]
//     = wscale * s[b,cin] * weight[cout,cin,tap]
//   with m=l&15, koff = j + ((l>>4)<<4), k = kc*32+koff,
//        tap = k/Cin, cin = k%Cin, cout = ct*16+m.
// Done ONCE per conv instead of once per conv block.
// ---------------------------------------------------------------------------
__global__ void __launch_bounds__(256)
build_wfrag_kernel(const float* __restrict__ weight, const float* __restrict__ s,
                   _Float16* __restrict__ wfrag, int Cin, int Cout, float wscale)
{
    const int CT = Cout >> 4;              // cout tiles
    const int NC = (9 * Cin) >> 5;         // K chunks
    const int perB = CT * NC * 512;
    const int total = 8 * perB;
    const int i = blockIdx.x * 256 + threadIdx.x;
    if (i >= total) return;

    const int b   = i / perB;
    int r         = i - b * perB;
    const int ct  = r / (NC * 512);
    r            -= ct * (NC * 512);
    const int kc  = r >> 9;
    const int rem = r & 511;
    const int l   = rem >> 4;
    const int j   = rem & 15;
    const int m   = l & 15;
    const int koff = j + ((l >> 4) << 4);
    const int k   = kc * 32 + koff;
    const int tap = k / Cin;
    const int cin = k - tap * Cin;
    const int cout = ct * 16 + m;
    wfrag[i] = (_Float16)(wscale * s[b * Cin + cin] *
                          weight[((size_t)cout * Cin + cin) * 9 + tap]);
}

// ---------------------------------------------------------------------------
// Modulated 3x3 conv, implicit GEMM, WMMA f32_16x16x32_f16, NHWC activations.
//   M = 16 couts, N = 32 pixels/wave, K = 9*Cin tap-major.
//   A fragment: one coalesced 32B v16h global load per chunk (precomputed,
//               L2-resident). No LDS, no barriers -> high occupancy.
//   B fragment: one aligned 32B contiguous v16h global load per lane (NHWC).
//   Epilogue: demod + noise + lrelu, packed v8h (16B) NHWC stores.
// Block = 128 threads (4 waves). EXEC is full at every WMMA.
// ---------------------------------------------------------------------------
__global__ void __launch_bounds__(128)
modconv_wmma(const _Float16* __restrict__ x, const _Float16* __restrict__ wfrag,
             const float* __restrict__ dmod, const float* __restrict__ noiseW,
             const float* __restrict__ noise, _Float16* __restrict__ out,
             int Cin, int Cout, int Hin, int Win, int Hout, int Wout,
             int stride, int xBatchStride)
{
    const int b = blockIdx.z;
    const int ct = blockIdx.y;
    const int cout0 = ct * 16;

    const int lane = threadIdx.x & 31;
    const int wave = threadIdx.x >> 5;
    const int hi   = lane >> 4;
    const int n    = lane & 15;
    const int p0   = (blockIdx.x * 4 + wave) * 32;      // 32 pixels per wave
    const int y    = p0 / Wout;                          // uniform per wave
    const int xA   = (p0 - y * Wout) + n;                // tile 0 column
    const int xB   = xA + 16;                            // tile 1 column

    const int nChunks = (9 * Cin) >> 5;
    const int CT = Cout >> 4;
    const _Float16* afrag = wfrag + ((size_t)(b * CT + ct) * nChunks) * 512 + lane * 16;
    const _Float16* xbase = x + (size_t)b * xBatchStride;

    v8f c0 = {}, c1 = {};
    int cin0 = 0, dy = -1, dx = -1;          // incremental tap/cin tracking
    for (int kc = 0; kc < nChunks; ++kc) {
        const int iy  = y * stride + dy;
        const int ix0 = xA * stride + dx;
        const int ix1 = xB * stride + dx;

        const v16h a = *(const v16h*)afrag;
        afrag += 512;

        const int cinOff = cin0 + (hi << 4);
        const _Float16* p0ptr = &xbase[((size_t)iy * Win + ix0) * Cin + cinOff];
        const _Float16* p1ptr = &xbase[((size_t)iy * Win + ix1) * Cin + cinOff];
        const bool okY = (iy >= 0) & (iy < Hin);
        v16h b0 = {}, b1 = {};
        if (okY & (ix0 >= 0) & (ix0 < Win)) b0 = *(const v16h*)p0ptr;
        if (okY & (ix1 >= 0) & (ix1 < Win)) b1 = *(const v16h*)p1ptr;

        __builtin_prefetch(p0ptr + 32, 0, 0);   // speculative: next cin block

        c0 = __builtin_amdgcn_wmma_f32_16x16x32_f16(false, a, false, b0,
                                                    (short)0, c0, false, false);
        c1 = __builtin_amdgcn_wmma_f32_16x16x32_f16(false, a, false, b1,
                                                    (short)0, c1, false, false);

        cin0 += 32;
        if (cin0 == Cin) {                   // advance to next 3x3 tap
            cin0 = 0;
            if (++dx > 1) { dx = -1; ++dy; }
        }
    }

    // Epilogue: demodulate, add noise, leaky ReLU, packed NHWC f16 stores.
    const float nz0 = noise[((size_t)b * Hout + y) * Wout + xA];
    const float nz1 = noise[((size_t)b * Hout + y) * Wout + xB];
    v8h h0, h1;
#pragma unroll
    for (int r = 0; r < 8; ++r) {
        const int cout = cout0 + (hi << 3) + r;
        const float dm = dmod[b * Cout + cout];
        const float nw = noiseW[cout];
        float v0 = c0[r] * dm + nw * nz0;
        float v1 = c1[r] * dm + nw * nz1;
        v0 = (v0 >= 0.f) ? v0 : 0.2f * v0;
        v1 = (v1 >= 0.f) ? v1 : 0.2f * v1;
        h0[r] = (_Float16)v0;
        h1[r] = (_Float16)v1;
    }
    const int coutOff = cout0 + (hi << 3);
    *(v8h*)&out[(((size_t)b * Hout + y) * Wout + xA) * Cout + coutOff] = h0;
    *(v8h*)&out[(((size_t)b * Hout + y) * Wout + xB) * Cout + coutOff] = h1;
}

// ---------------------------------------------------------------------------
// Bilinear resize (32x32 -> 256x240, half-pixel centers) fused with 1x1
// to-gray modulated conv (no demod). NHWC f16 input, f32 output.
// ---------------------------------------------------------------------------
__global__ void __launch_bounds__(256)
rgb_resize_kernel(const _Float16* __restrict__ xin,   // (B,32,32,32c) NHWC f16
                  const float* __restrict__ s,        // (B,32)
                  const float* __restrict__ rgbw,     // (32)
                  float* __restrict__ outp)           // (B,1,256,240)
{
    const int b = blockIdx.y;
    __shared__ float coeff[32];
    if (threadIdx.x < 32)
        coeff[threadIdx.x] = 0.17677669529663687f /* 1/sqrt(32) */ *
                             s[b * 32 + threadIdx.x] * rgbw[threadIdx.x];
    __syncthreads();

    const int p = blockIdx.x * 256 + threadIdx.x;     // 0 .. 61439
    const int y = p / 240;
    const int xq = p - y * 240;
    const float sy = (y + 0.5f) * (32.f / 256.f) - 0.5f;
    const float sx = (xq + 0.5f) * (32.f / 240.f) - 0.5f;
    const int y0 = (int)floorf(sy);
    const int x0 = (int)floorf(sx);
    const float fy = sy - (float)y0;
    const float fx = sx - (float)x0;
    const int y0c = min(max(y0, 0), 31), y1c = min(max(y0 + 1, 0), 31);
    const int x0c = min(max(x0, 0), 31), x1c = min(max(x0 + 1, 0), 31);
    const float w00 = (1.f - fy) * (1.f - fx), w01 = (1.f - fy) * fx;
    const float w10 = fy * (1.f - fx),         w11 = fy * fx;

    const _Float16* base = xin + (size_t)b * 32 * 32 * 32;
    const int p00 = (y0c * 32 + x0c) * 32, p01 = (y0c * 32 + x1c) * 32;
    const int p10 = (y1c * 32 + x0c) * 32, p11 = (y1c * 32 + x1c) * 32;

    float acc = 0.f;
    for (int c = 0; c < 32; ++c) {
        const float v = w00 * (float)base[p00 + c] + w01 * (float)base[p01 + c] +
                        w10 * (float)base[p10 + c] + w11 * (float)base[p11 + c];
        acc += coeff[c] * v;
    }
    outp[((size_t)b * 256 + y) * 240 + xq] = acc;
}

// ---------------------------------------------------------------------------
extern "C" void kernel_launch(void* const* d_in, const int* in_sizes, int n_in,
                              void* d_out, int out_size, void* d_ws, size_t ws_size,
                              hipStream_t stream)
{
    (void)in_sizes; (void)n_in; (void)out_size; (void)ws_size;
    const float* z       = (const float*)d_in[0];
    const int*   labels  = (const int*)  d_in[1];
    const float* emb     = (const float*)d_in[2];
    const float* map_W   = (const float*)d_in[3];
    const float* map_b   = (const float*)d_in[4];
    const float* map_g   = (const float*)d_in[5];
    const float* map_be  = (const float*)d_in[6];
    const float* constw  = (const float*)d_in[7];
    const float* b1c1_w  = (const float*)d_in[8];
    const float* b1c1_m  = (const float*)d_in[9];
    const float* b1c2_w  = (const float*)d_in[10];
    const float* b1c2_m  = (const float*)d_in[11];
    const float* b1n1_w  = (const float*)d_in[12];
    const float* b1n2_w  = (const float*)d_in[13];
    const float* b2c1_w  = (const float*)d_in[14];
    const float* b2c1_m  = (const float*)d_in[15];
    const float* b2c2_w  = (const float*)d_in[16];
    const float* b2c2_m  = (const float*)d_in[17];
    const float* b2n1_w  = (const float*)d_in[18];
    const float* b2n2_w  = (const float*)d_in[19];
    const float* b3c1_w  = (const float*)d_in[20];
    const float* b3c1_m  = (const float*)d_in[21];
    const float* b3c2_w  = (const float*)d_in[22];
    const float* b3c2_m  = (const float*)d_in[23];
    const float* b3n1_w  = (const float*)d_in[24];
    const float* b3n2_w  = (const float*)d_in[25];
    const float* rgb_w   = (const float*)d_in[26];
    const float* rgb_m   = (const float*)d_in[27];
    const float* n1a     = (const float*)d_in[28];
    const float* n1b     = (const float*)d_in[29];
    const float* n2a     = (const float*)d_in[30];
    const float* n2b     = (const float*)d_in[31];
    const float* n3a     = (const float*)d_in[32];
    const float* n3b     = (const float*)d_in[33];
    float* outp = (float*)d_out;

    // ---- workspace carve-up ----
    char* ws = (char*)d_ws;
    size_t off = 0;
    auto carve = [&](size_t bytes) -> char* {
        char* p = ws + off;
        off = (off + bytes + 255) & ~(size_t)255;
        return p;
    };
    float* wlat = (float*)carve(8 * 512 * 4);
    float* s1a  = (float*)carve(8 * 128 * 4);
    float* s1b  = (float*)carve(8 * 128 * 4);
    float* s2a  = (float*)carve(8 * 128 * 4);
    float* s2b  = (float*)carve(8 * 64 * 4);
    float* s3a  = (float*)carve(8 * 64 * 4);
    float* s3b  = (float*)carve(8 * 32 * 4);
    float* srgb = (float*)carve(8 * 32 * 4);
    float* d1a  = (float*)carve(8 * 128 * 4);
    float* d1b  = (float*)carve(8 * 128 * 4);
    float* d2a  = (float*)carve(8 * 64 * 4);
    float* d2b  = (float*)carve(8 * 64 * 4);
    float* d3a  = (float*)carve(8 * 32 * 4);
    float* d3b  = (float*)carve(8 * 32 * 4);
    // Fragment-swizzled modulated weights, per conv: 8 * Cout * 9*Cin f16
    _Float16* wf1a = (_Float16*)carve((size_t)8 * 128 * 1152 * 2);
    _Float16* wf1b = (_Float16*)carve((size_t)8 * 128 * 1152 * 2);
    _Float16* wf2a = (_Float16*)carve((size_t)8 * 64 * 1152 * 2);
    _Float16* wf2b = (_Float16*)carve((size_t)8 * 64 * 576 * 2);
    _Float16* wf3a = (_Float16*)carve((size_t)8 * 32 * 576 * 2);
    _Float16* wf3b = (_Float16*)carve((size_t)8 * 32 * 288 * 2);
    _Float16* constH = (_Float16*)carve((size_t)128 * 256 * 256 * 2);  // HWC
    _Float16* bufA   = (_Float16*)carve((size_t)8 * 128 * 256 * 256 * 2);
    _Float16* bufB   = (_Float16*)carve((size_t)8 * 128 * 128 * 128 * 2);

    // ---- mapping network ----
    mapping_kernel<<<1, 512, 0, stream>>>(z, labels, emb, map_W, map_b, map_g, map_be, wlat);

    // ---- styles ----
    style_kernel<<<dim3(1, 8), 128, 0, stream>>>(wlat, b1c1_m, s1a, 128);
    style_kernel<<<dim3(1, 8), 128, 0, stream>>>(wlat, b1c2_m, s1b, 128);
    style_kernel<<<dim3(1, 8), 128, 0, stream>>>(wlat, b2c1_m, s2a, 128);
    style_kernel<<<dim3(1, 8), 128, 0, stream>>>(wlat, b2c2_m, s2b, 64);
    style_kernel<<<dim3(1, 8), 128, 0, stream>>>(wlat, b3c1_m, s3a, 64);
    style_kernel<<<dim3(1, 8), 128, 0, stream>>>(wlat, b3c2_m, s3b, 32);
    style_kernel<<<dim3(1, 8), 128, 0, stream>>>(wlat, rgb_m,  srgb, 32);

    // ---- demodulation coefficients ----
    const float sc128 = 1.0f / sqrtf(128.f * 9.f);
    const float sc64  = 1.0f / sqrtf(64.f * 9.f);
    const float sc32  = 1.0f / sqrtf(32.f * 9.f);
    demod_kernel<<<dim3(128, 8), 64, 0, stream>>>(s1a, b1c1_w, d1a, 128, 128, sc128);
    demod_kernel<<<dim3(128, 8), 64, 0, stream>>>(s1b, b1c2_w, d1b, 128, 128, sc128);
    demod_kernel<<<dim3(64, 8),  64, 0, stream>>>(s2a, b2c1_w, d2a, 128, 64,  sc128);
    demod_kernel<<<dim3(64, 8),  64, 0, stream>>>(s2b, b2c2_w, d2b, 64,  64,  sc64);
    demod_kernel<<<dim3(32, 8),  64, 0, stream>>>(s3a, b3c1_w, d3a, 64,  32,  sc64);
    demod_kernel<<<dim3(32, 8),  64, 0, stream>>>(s3b, b3c2_w, d3b, 32,  32,  sc32);

    // ---- modulated, fragment-swizzled weights (once per conv) ----
    build_wfrag_kernel<<<(8 * 128 * 1152 + 255) / 256, 256, 0, stream>>>(b1c1_w, s1a, wf1a, 128, 128, sc128);
    build_wfrag_kernel<<<(8 * 128 * 1152 + 255) / 256, 256, 0, stream>>>(b1c2_w, s1b, wf1b, 128, 128, sc128);
    build_wfrag_kernel<<<(8 * 64 * 1152 + 255) / 256, 256, 0, stream>>>(b2c1_w, s2a, wf2a, 128, 64, sc128);
    build_wfrag_kernel<<<(8 * 64 * 576 + 255) / 256, 256, 0, stream>>>(b2c2_w, s2b, wf2b, 64, 64, sc64);
    build_wfrag_kernel<<<(8 * 32 * 576 + 255) / 256, 256, 0, stream>>>(b3c1_w, s3a, wf3a, 64, 32, sc64);
    build_wfrag_kernel<<<(8 * 32 * 288 + 255) / 256, 256, 0, stream>>>(b3c2_w, s3b, wf3b, 32, 32, sc32);

    // ---- const CHW f32 -> HWC f16 ----
    {
        const int n = 128 * 256 * 256;
        chw_to_hwc_f16<<<n / 256, 256, 0, stream>>>(constw, constH, 128, 256 * 256);
    }

    // ---- block 1 ----  (all activations NHWC f16)
    modconv_wmma<<<dim3(512, 8, 8), 128, 0, stream>>>(
        constH, wf1a, d1a, b1n1_w, n1a, bufA,
        128, 128, 256, 256, 256, 256, 1, 0);
    modconv_wmma<<<dim3(128, 8, 8), 128, 0, stream>>>(
        bufA, wf1b, d1b, b1n2_w, n1b, bufB,
        128, 128, 256, 256, 128, 128, 2, 256 * 256 * 128);
    // ---- block 2 ----
    modconv_wmma<<<dim3(128, 4, 8), 128, 0, stream>>>(
        bufB, wf2a, d2a, b2n1_w, n2a, bufA,
        128, 64, 128, 128, 128, 128, 1, 128 * 128 * 128);
    modconv_wmma<<<dim3(32, 4, 8), 128, 0, stream>>>(
        bufA, wf2b, d2b, b2n2_w, n2b, bufB,
        64, 64, 128, 128, 64, 64, 2, 128 * 128 * 64);
    // ---- block 3 ----
    modconv_wmma<<<dim3(32, 2, 8), 128, 0, stream>>>(
        bufB, wf3a, d3a, b3n1_w, n3a, bufA,
        64, 32, 64, 64, 64, 64, 1, 64 * 64 * 64);
    modconv_wmma<<<dim3(8, 2, 8), 128, 0, stream>>>(
        bufA, wf3b, d3b, b3n2_w, n3b, bufB,
        32, 32, 64, 64, 32, 32, 2, 64 * 64 * 32);

    // ---- bilinear resize + 1x1 to-gray conv ----
    rgb_resize_kernel<<<dim3(240, 8), 256, 0, stream>>>(bufB, srgb, rgb_w, outp);
}
